// DecisionTransformer_17987323036311
// MI455X (gfx1250) — matrix operand order repaired and checked by reading, not compiled, not run on hardware
//
#include <hip/hip_runtime.h>
#include <hip/hip_bf16.h>
#include <math.h>

// ---------------- problem constants ----------------
#define HD     768
#define FD     3072
#define NHEAD  12
#define DHEAD  64
#define NB     32
#define NT     128
#define SPAD   416          // 384 tokens padded to 26*16 (expert stage uses 404 valid)
#define MTOT   (NB*SPAD)    // 13312 rows
#define SDIM   17
#define ADIM   6
#define TOKN   20
#define NTASK  3
#define ATT_SCALE 0.125f    // 1/sqrt(64)

#define USE_ASYNC_LDS 1     // gfx1250 global_load_async_to_lds_b128 staging of GEMM B tiles

typedef __attribute__((ext_vector_type(16))) __bf16 v16bf;
typedef __attribute__((ext_vector_type(8)))  __bf16 v8bf;
typedef __attribute__((ext_vector_type(8)))  float  v8f;

// ---------------- WMMA fragment helpers (CDNA5 16x16x32 bf16) ----------------
// A-matrix (16xK=32): lane r=lane&15 holds row r; half=lane>>4 selects K groups
// {half*8..half*8+7} (vec idx 0..7) and {16+half*8..} (vec idx 8..15).  (ISA 7.12.2)
static __device__ inline v16bf frag_a(const __bf16* p, int ld) {
  int lane = threadIdx.x & 31;
  int half = lane >> 4, r = lane & 15;
  const __bf16* q = p + (size_t)r * ld + half * 8;
  v8bf lo = *(const v8bf*)q;
  v8bf hi = *(const v8bf*)(q + 16);
  return __builtin_shufflevector(lo, hi, 0,1,2,3,4,5,6,7,8,9,10,11,12,13,14,15);
}
// B-matrix (K=32 x 16): lane n=lane&15 holds column n; half=lane>>4 selects K groups
// {half*8..half*8+7} (vec idx 0..7) and {16+half*8..} (vec idx 8..15), matching frag_a.
static __device__ inline v16bf frag_b(const __bf16* p, int ld) {
  int lane = threadIdx.x & 31;
  int half = lane >> 4, n = lane & 15;
  const __bf16* q = p + (size_t)n * ld + half * 8;
  v8bf lo = *(const v8bf*)q;
  v8bf hi = *(const v8bf*)(q + 16);
  return __builtin_shufflevector(lo, hi, 0,1,2,3,4,5,6,7,8,9,10,11,12,13,14,15);
}
static __device__ inline v8f wmma_bf16(v16bf a, v16bf b, v8f c) {
  return __builtin_amdgcn_wmma_f32_16x16x32_bf16(false, a, false, b, (short)0, c, false, false);
}

#if USE_ASYNC_LDS
// LDS byte offset of a __shared__ object (addrspacecast generic->AS3, then ptrtoint)
static __device__ inline unsigned to_lds_off(const void* p) {
  return (unsigned)(unsigned long long)(__attribute__((address_space(3))) const void*)p;
}
// gfx1250 async DMA: 16 bytes global -> LDS per active lane (tracked by ASYNCcnt)
static __device__ inline void gl2lds_b128(unsigned ldsoff, const void* g) {
  asm volatile("global_load_async_to_lds_b128 %0, %1, off"
               :: "v"(ldsoff), "v"((unsigned long long)g)
               : "memory");
}
static __device__ inline void wait_asynccnt0() {
  asm volatile("s_wait_asynccnt 0x0" ::: "memory");
}
#endif

// ---------------- weight transpose + fp32->bf16 convert ----------------
// W: (K,N) fp32 row-major  ->  WT: (N,K) bf16 row-major
__global__ __launch_bounds__(256)
void dt_wtrans_kernel(const float* __restrict__ W, __bf16* __restrict__ WT, int K, int N) {
  __shared__ float tile[32][33];
  int tx = threadIdx.x & 31, ty = threadIdx.x >> 5; // 32x8
  int nbase = blockIdx.x * 32, kbase = blockIdx.y * 32;
#pragma unroll
  for (int j = 0; j < 32; j += 8) {
    int kk = kbase + ty + j, nn = nbase + tx;
    tile[ty + j][tx] = (kk < K && nn < N) ? W[(size_t)kk * N + nn] : 0.0f;
  }
  __syncthreads();
#pragma unroll
  for (int j = 0; j < 32; j += 8) {
    int nn = nbase + ty + j, kk = kbase + tx;
    if (nn < N && kk < K) WT[(size_t)nn * K + kk] = (__bf16)tile[tx][ty + j];
  }
}

// ---------------- main WMMA GEMM: out = act(A*W + bias) (+resid) ----------------
// A: (M,K) bf16;  WT: (N,K) bf16 (pre-transposed);  act: 0=none, 1=exact gelu
// Block: 256 threads = 8 waves, each wave owns a 32x64 output tile (block: 256x64).
// B tile (64x32 bf16, shared by all waves) is double-buffer staged in LDS with
// gfx1250 async global->LDS DMA so the next K-step's fetch overlaps the WMMAs.
__global__ __launch_bounds__(256)
void dt_gemm_kernel(const __bf16* __restrict__ A, const __bf16* __restrict__ WT,
                    const float* __restrict__ bias, const float* __restrict__ resid,
                    float* __restrict__ outF, __bf16* __restrict__ outB,
                    int M, int N, int K, int act) {
#if USE_ASYNC_LDS
  __shared__ __align__(16) __bf16 Bt[2][64 * 32];
#endif
  int wid = threadIdx.x >> 5;
  int lane = threadIdx.x & 31;
  int row0 = (blockIdx.y * 8 + wid) * 32;
  if (row0 + 32 > M) row0 = M - 32;   // defensive (grid exactly covers M here)
  int n0 = blockIdx.x * 64;

  v8f acc[2][4];
#pragma unroll
  for (int i = 0; i < 2; i++)
#pragma unroll
    for (int t = 0; t < 4; t++) acc[i][t] = (v8f){};

  const __bf16* arow0 = A + (size_t)row0 * K;
  const __bf16* arow1 = arow0 + (size_t)16 * K;

#if USE_ASYNC_LDS
  // staging map: thread i copies 16B: row = i>>2 (0..63), chunk = (i&3)*8
  int srow = threadIdx.x >> 2;
  int schunk = (threadIdx.x & 3) * 8;
  const __bf16* sgl = WT + (size_t)(n0 + srow) * K + schunk;
  unsigned sdst0 = to_lds_off(&Bt[0][srow * 32 + schunk]);
  unsigned sdst1 = to_lds_off(&Bt[1][srow * 32 + schunk]);
  gl2lds_b128(sdst0, sgl);            // k0 = 0 tile -> buf 0
  wait_asynccnt0();
  __syncthreads();
  int buf = 0;
  for (int k0 = 0; k0 < K; k0 += 32) {
    if (k0 + 32 < K)                  // kick off next tile into the other buffer
      gl2lds_b128(buf ? sdst0 : sdst1, sgl + k0 + 32);
    v16bf a0 = frag_a(arow0 + k0, K);
    v16bf a1 = frag_a(arow1 + k0, K);
    if (k0 + 64 < K) {                // prefetch A stream two K-steps ahead
      __builtin_prefetch(arow0 + k0 + 64, 0, 1);
      __builtin_prefetch(arow1 + k0 + 64, 0, 1);
    }
    const __bf16* bbase = &Bt[buf][0];
#pragma unroll
    for (int t = 0; t < 4; t++) {
      v16bf b = frag_b(bbase + t * 16 * 32, 32);
      acc[0][t] = wmma_bf16(a0, b, acc[0][t]);
      acc[1][t] = wmma_bf16(a1, b, acc[1][t]);
    }
    wait_asynccnt0();                 // next tile landed (issued before compute)
    __syncthreads();
    buf ^= 1;
  }
#else
  const __bf16* w0 = WT + (size_t)(n0 +  0) * K;
  const __bf16* w1 = WT + (size_t)(n0 + 16) * K;
  const __bf16* w2 = WT + (size_t)(n0 + 32) * K;
  const __bf16* w3 = WT + (size_t)(n0 + 48) * K;
  for (int k0 = 0; k0 < K; k0 += 32) {
    v16bf a0 = frag_a(arow0 + k0, K);
    v16bf a1 = frag_a(arow1 + k0, K);
    v16bf b0 = frag_b(w0 + k0, K);
    acc[0][0] = wmma_bf16(a0, b0, acc[0][0]); acc[1][0] = wmma_bf16(a1, b0, acc[1][0]);
    v16bf b1 = frag_b(w1 + k0, K);
    acc[0][1] = wmma_bf16(a0, b1, acc[0][1]); acc[1][1] = wmma_bf16(a1, b1, acc[1][1]);
    v16bf b2 = frag_b(w2 + k0, K);
    acc[0][2] = wmma_bf16(a0, b2, acc[0][2]); acc[1][2] = wmma_bf16(a1, b2, acc[1][2]);
    v16bf b3 = frag_b(w3 + k0, K);
    acc[0][3] = wmma_bf16(a0, b3, acc[0][3]); acc[1][3] = wmma_bf16(a1, b3, acc[1][3]);
  }
#endif

  int half = lane >> 4, n = lane & 15;
#pragma unroll
  for (int i = 0; i < 2; i++) {
#pragma unroll
    for (int t = 0; t < 4; t++) {
      int col = n0 + t * 16 + n;
      float bv = bias ? bias[col] : 0.0f;
#pragma unroll
      for (int r = 0; r < 8; r++) {
        int row = row0 + i * 16 + half * 8 + r;
        float v = acc[i][t][r] + bv;
        if (act == 1) v = 0.5f * v * (1.0f + erff(v * 0.70710678118654752f));
        if (resid) v += resid[(size_t)row * N + col];
        if (outF) outF[(size_t)row * N + col] = v;
        if (outB) outB[(size_t)row * N + col] = (__bf16)v;
      }
    }
  }
}

// ---------------- V transpose: (B*SPAD,H) bf16 -> (B,NH,DHEAD,SPAD) bf16 ----------------
__global__ __launch_bounds__(256)
void dt_vtrans_kernel(const __bf16* __restrict__ v, __bf16* __restrict__ vt) {
  size_t idx = (size_t)blockIdx.x * 256 + threadIdx.x;
  size_t total = (size_t)NB * NHEAD * DHEAD * SPAD;
  if (idx >= total) return;
  int s = (int)(idx % SPAD); size_t rest = idx / SPAD;
  int d = (int)(rest % DHEAD); rest /= DHEAD;
  int h = (int)(rest % NHEAD); int b = (int)(rest / NHEAD);
  vt[idx] = v[(size_t)(b * SPAD + s) * HD + h * DHEAD + d];
}

// ---------------- flash attention (one wave per (b,head,16-query tile)) ----------------
__global__ __launch_bounds__(32)
void dt_attn_kernel(const __bf16* __restrict__ q, const __bf16* __restrict__ k,
                    const __bf16* __restrict__ vt, __bf16* __restrict__ outB, int Sact) {
  __shared__ __align__(16) __bf16 P[16 * 32];
  int b = blockIdx.z, h = blockIdx.y, it = blockIdx.x;
  int lane = threadIdx.x;
  int half = lane >> 4, nl = lane & 15;

  const __bf16* qp = q + (size_t)(b * SPAD + it * 16) * HD + h * DHEAD;
  v16bf aq0 = frag_a(qp, HD);        // d = 0..31
  v16bf aq1 = frag_a(qp + 32, HD);   // d = 32..63

  v8f o0 = {}, o1 = {}, o2 = {}, o3 = {};
  float mrow[8], lrow[8];
#pragma unroll
  for (int r = 0; r < 8; r++) { mrow[r] = -1e30f; lrow[r] = 0.0f; }

  int jmax = min(it * 16 + 15, Sact - 1);
  int njb = jmax / 32 + 1;
  for (int jb = 0; jb < njb; jb++) {
    int j0 = jb * 32;
    // scores for 32 keys (two 16-key tiles): S = Q*K^T (keys load directly as B-frags)
    v8f s0 = {}, s1 = {};
    {
      const __bf16* kp = k + (size_t)(b * SPAD + j0) * HD + h * DHEAD;
      s0 = wmma_bf16(aq0, frag_b(kp, HD), s0);
      s0 = wmma_bf16(aq1, frag_b(kp + 32, HD), s0);
      kp += (size_t)16 * HD;
      s1 = wmma_bf16(aq0, frag_b(kp, HD), s1);
      s1 = wmma_bf16(aq1, frag_b(kp + 32, HD), s1);
    }
    float st0[8], st1[8];
#pragma unroll
    for (int r = 0; r < 8; r++) {
      int qg = it * 16 + half * 8 + r;
      int jg0 = j0 + nl, jg1 = j0 + 16 + nl;
      st0[r] = (jg0 <= qg && jg0 < Sact) ? s0[r] * ATT_SCALE : -1e30f;
      st1[r] = (jg1 <= qg && jg1 < Sact) ? s1[r] * ATT_SCALE : -1e30f;
    }
#pragma unroll
    for (int r = 0; r < 8; r++) {
      float mx = fmaxf(st0[r], st1[r]);
      mx = fmaxf(mx, __shfl_xor(mx, 1, 32));
      mx = fmaxf(mx, __shfl_xor(mx, 2, 32));
      mx = fmaxf(mx, __shfl_xor(mx, 4, 32));
      mx = fmaxf(mx, __shfl_xor(mx, 8, 32));
      float mnew = fmaxf(mrow[r], mx);
      float alpha = __expf(mrow[r] - mnew);
      mrow[r] = mnew;
      float p0 = __expf(st0[r] - mnew);
      float p1 = __expf(st1[r] - mnew);
      float rs = p0 + p1;
      rs += __shfl_xor(rs, 1, 32);
      rs += __shfl_xor(rs, 2, 32);
      rs += __shfl_xor(rs, 4, 32);
      rs += __shfl_xor(rs, 8, 32);
      lrow[r] = lrow[r] * alpha + rs;
      o0[r] *= alpha; o1[r] *= alpha; o2[r] *= alpha; o3[r] *= alpha;
      P[(half * 8 + r) * 32 + nl] = (__bf16)p0;
      P[(half * 8 + r) * 32 + 16 + nl] = (__bf16)p1;
    }
    __syncthreads();
    // O += P * V  (P from LDS as A-frag, V columns contiguous via VT)
    v16bf pa = frag_a((const __bf16*)P, 32);
    const __bf16* vp = vt + ((size_t)(b * NHEAD + h) * DHEAD) * SPAD + j0;
    o0 = wmma_bf16(pa, frag_b(vp +  0 * SPAD, SPAD), o0);
    o1 = wmma_bf16(pa, frag_b(vp + 16 * SPAD, SPAD), o1);
    o2 = wmma_bf16(pa, frag_b(vp + 32 * SPAD, SPAD), o2);
    o3 = wmma_bf16(pa, frag_b(vp + 48 * SPAD, SPAD), o3);
    __syncthreads();
  }
#pragma unroll
  for (int r = 0; r < 8; r++) {
    float inv = 1.0f / lrow[r];
    size_t row = (size_t)(b * SPAD + it * 16 + half * 8 + r) * HD + h * DHEAD;
    outB[row +  0 + nl] = (__bf16)(o0[r] * inv);
    outB[row + 16 + nl] = (__bf16)(o1[r] * inv);
    outB[row + 32 + nl] = (__bf16)(o2[r] * inv);
    outB[row + 48 + nl] = (__bf16)(o3[r] * inv);
  }
}

// ---------------- layernorm (one row per block), writes fp32 + bf16 ----------------
__global__ __launch_bounds__(256)
void dt_ln_kernel(const float* __restrict__ x, const float* __restrict__ sc,
                  const float* __restrict__ bi, float* __restrict__ yF,
                  __bf16* __restrict__ yB) {
  __shared__ float red[256];
  int tid = threadIdx.x;
  const float* xr = x + (size_t)blockIdx.x * HD;
  float v0 = xr[tid], v1 = xr[tid + 256], v2 = xr[tid + 512];
  red[tid] = v0 + v1 + v2;
  __syncthreads();
  for (int o = 128; o > 0; o >>= 1) { if (tid < o) red[tid] += red[tid + o]; __syncthreads(); }
  float mean = red[0] * (1.0f / HD);
  __syncthreads();
  float d0 = v0 - mean, d1 = v1 - mean, d2 = v2 - mean;
  red[tid] = d0 * d0 + d1 * d1 + d2 * d2;
  __syncthreads();
  for (int o = 128; o > 0; o >>= 1) { if (tid < o) red[tid] += red[tid + o]; __syncthreads(); }
  float inv = rsqrtf(red[0] * (1.0f / HD) + 1e-5f);
  size_t ro = (size_t)blockIdx.x * HD;
  float r0 = d0 * inv * sc[tid] + bi[tid];
  float r1 = d1 * inv * sc[tid + 256] + bi[tid + 256];
  float r2 = d2 * inv * sc[tid + 512] + bi[tid + 512];
  yF[ro + tid] = r0;        yB[ro + tid] = (__bf16)r0;
  yF[ro + tid + 256] = r1;  yB[ro + tid + 256] = (__bf16)r1;
  yF[ro + tid + 512] = r2;  yB[ro + tid + 512] = (__bf16)r2;
}

// ---------------- embeddings: rows 3t+{0,1,2} = re,se,ae ----------------
__global__ __launch_bounds__(256)
void dt_embed_kernel(const float* __restrict__ states, const float* __restrict__ actions,
                     const float* __restrict__ rtg, const int* __restrict__ tsteps,
                     const float* __restrict__ smean, const float* __restrict__ sstd,
                     const float* __restrict__ Wt, const float* __restrict__ Ws,
                     const float* __restrict__ bs, const float* __restrict__ Wa,
                     const float* __restrict__ ba, const float* __restrict__ Wr,
                     const float* __restrict__ br, float* __restrict__ hF) {
  __shared__ float sh[24];
  int bt = blockIdx.x;
  int b = bt / NT, t = bt % NT;
  if (threadIdx.x < SDIM)
    sh[threadIdx.x] = (states[(size_t)bt * SDIM + threadIdx.x] - smean[threadIdx.x]) / sstd[threadIdx.x];
  else if (threadIdx.x < SDIM + ADIM)
    sh[threadIdx.x] = actions[(size_t)bt * ADIM + (threadIdx.x - SDIM)];
  else if (threadIdx.x == SDIM + ADIM)
    sh[23] = rtg[bt] * (1.0f / 1000.0f);
  __syncthreads();
  int ts = tsteps[bt];
  float* hr = hF + (size_t)(b * SPAD + 3 * t) * HD;
  for (int c = threadIdx.x; c < HD; c += 256) {
    float te = Wt[(size_t)ts * HD + c];
    float se = bs[c] + te;
#pragma unroll
    for (int i = 0; i < SDIM; i++) se += sh[i] * Ws[(size_t)i * HD + c];
    float ae = ba[c] + te;
#pragma unroll
    for (int i = 0; i < ADIM; i++) ae += sh[SDIM + i] * Wa[(size_t)i * HD + c];
    float re = br[c] + te + sh[23] * Wr[c];
    hr[0 * HD + c] = re;
    hr[1 * HD + c] = se;
    hr[2 * HD + c] = ae;
  }
}

__global__ __launch_bounds__(256)
void dt_pad_kernel(float* __restrict__ hF) {
  int b = blockIdx.y, rr = blockIdx.x;
  size_t ro = (size_t)(b * SPAD + 3 * NT + rr) * HD;
  for (int c = threadIdx.x; c < HD; c += 256) hF[ro + c] = 0.0f;
}

// expert prompt tokens into rows 384..415 (>=404 zeroed), fp32 + bf16
__global__ __launch_bounds__(256)
void dt_tok_kernel(const float* __restrict__ tokE, const int* __restrict__ task,
                   float* __restrict__ hF, __bf16* __restrict__ hB) {
  int b = blockIdx.y, rr = blockIdx.x;
  const float* src = tokE + (size_t)task[0] * TOKN * HD + (size_t)rr * HD;
  size_t ro = (size_t)(b * SPAD + 3 * NT + rr) * HD;
  for (int c = threadIdx.x; c < HD; c += 256) {
    float v = (rr < TOKN) ? src[c] : 0.0f;
    hF[ro + c] = v;
    hB[ro + c] = (__bf16)v;
  }
}

// ---------------- output heads ----------------
__global__ __launch_bounds__(32)
void dt_head_kernel(const float* __restrict__ hF, const float* __restrict__ Wpa,
                    const float* __restrict__ bpa, const float* __restrict__ Wps,
                    const float* __restrict__ bps, const float* __restrict__ Wpr,
                    const float* __restrict__ bpr, const float* __restrict__ alow,
                    const float* __restrict__ ahigh, float* __restrict__ out) {
  int bt = blockIdx.x;
  int b = bt / NT, t = bt % NT;
  const float* h1 = hF + (size_t)(b * SPAD + 3 * t + 1) * HD; // state token -> action
  const float* h2 = hF + (size_t)(b * SPAD + 3 * t + 2) * HD; // action token -> sp, rp
  int j = threadIdx.x;
  float* sp = out;
  float* ap = out + (size_t)NB * NT * SDIM;
  float* rp = ap + (size_t)NB * NT * ADIM;
  if (j < SDIM) {
    float acc = bps[j];
    for (int c = 0; c < HD; c++) acc += h2[c] * Wps[(size_t)c * SDIM + j];
    sp[(size_t)bt * SDIM + j] = acc;
  } else if (j < SDIM + ADIM) {
    int jj = j - SDIM;
    float acc = bpa[jj];
    for (int c = 0; c < HD; c++) acc += h1[c] * Wpa[(size_t)c * ADIM + jj];
    float ta = tanhf(acc);
    ap[(size_t)bt * ADIM + jj] = alow[jj] + (ta + 1.0f) * (ahigh[jj] - alow[jj]) * 0.5f;
  } else if (j == SDIM + ADIM) {
    float acc = bpr[0];
    for (int c = 0; c < HD; c++) acc += h2[c] * Wpr[c];
    rp[bt] = acc;
  }
}

// ==================== host launcher ====================
extern "C" void kernel_launch(void* const* d_in, const int* in_sizes, int n_in,
                              void* d_out, int out_size, void* d_ws, size_t ws_size,
                              hipStream_t stream) {
  (void)in_sizes; (void)n_in; (void)out_size; (void)ws_size;
  // inputs in setup_inputs() dict (insertion) order, params flattened recursively
  const float* states  = (const float*)d_in[0];
  const float* actions = (const float*)d_in[1];
  const float* rtg     = (const float*)d_in[2];
  const int*   tsteps  = (const int*)d_in[3];
  const int*   task_id = (const int*)d_in[4];
  const float* smean = (const float*)d_in[5];
  const float* sstd  = (const float*)d_in[6];
  const float* alow  = (const float*)d_in[7];
  const float* ahigh = (const float*)d_in[8];
  const float* Wt = (const float*)d_in[9];
  const float* Ws = (const float*)d_in[10];
  const float* bs = (const float*)d_in[11];
  const float* Wa = (const float*)d_in[12];
  const float* ba = (const float*)d_in[13];
  const float* Wr = (const float*)d_in[14];
  const float* br = (const float*)d_in[15];
  const float* ln0s = (const float*)d_in[16];
  const float* ln0b = (const float*)d_in[17];
  // main blocks (stacked over 6 layers)
  const float* bWq = (const float*)d_in[18]; const float* bbq = (const float*)d_in[19];
  const float* bWk = (const float*)d_in[20]; const float* bbk = (const float*)d_in[21];
  const float* bWv = (const float*)d_in[22]; const float* bbv = (const float*)d_in[23];
  const float* bWo = (const float*)d_in[24]; const float* bbo = (const float*)d_in[25];
  const float* bW1 = (const float*)d_in[26]; const float* bb1 = (const float*)d_in[27];
  const float* bW2 = (const float*)d_in[28]; const float* bb2 = (const float*)d_in[29];
  const float* bl1s = (const float*)d_in[30]; const float* bl1b = (const float*)d_in[31];
  const float* bl2s = (const float*)d_in[32]; const float* bl2b = (const float*)d_in[33];
  // expert blocks (stacked over 2)
  const float* eWq = (const float*)d_in[34]; const float* ebq = (const float*)d_in[35];
  const float* eWk = (const float*)d_in[36]; const float* ebk = (const float*)d_in[37];
  const float* eWv = (const float*)d_in[38]; const float* ebv = (const float*)d_in[39];
  const float* eWo = (const float*)d_in[40]; const float* ebo = (const float*)d_in[41];
  const float* eW1 = (const float*)d_in[42]; const float* eb1 = (const float*)d_in[43];
  const float* eW2 = (const float*)d_in[44]; const float* eb2 = (const float*)d_in[45];
  const float* el1s = (const float*)d_in[46]; const float* el1b = (const float*)d_in[47];
  const float* el2s = (const float*)d_in[48]; const float* el2b = (const float*)d_in[49];
  const float* eTok = (const float*)d_in[50];
  const float* Wpa = (const float*)d_in[51]; const float* bpa = (const float*)d_in[52];
  const float* Wps = (const float*)d_in[53]; const float* bps = (const float*)d_in[54];
  const float* Wpr = (const float*)d_in[55]; const float* bpr = (const float*)d_in[56];

  // ---- workspace carve (~239 MB) ----
  char* wp = (char*)d_ws;
  auto take = [&](size_t bytes) { char* r = wp; wp += (bytes + 255) & ~(size_t)255; return r; };
  float*  hF   = (float*)take((size_t)MTOT * HD * 4);
  float*  resF = (float*)take((size_t)MTOT * HD * 4);
  __bf16* hB   = (__bf16*)take((size_t)MTOT * HD * 2);
  __bf16* kB   = (__bf16*)take((size_t)MTOT * HD * 2);
  __bf16* VT   = (__bf16*)take((size_t)NB * NHEAD * DHEAD * SPAD * 2);
  __bf16* big  = (__bf16*)take((size_t)MTOT * FD * 2);
  __bf16* qB = big;                        // aliases ff buffer (q consumed before MLP)
  __bf16* vB = big + (size_t)MTOT * HD;    // aliases ff buffer (v consumed by vtrans)
  __bf16* ffB = big;
  __bf16* WqT = (__bf16*)take((size_t)HD * HD * 2);
  __bf16* WkT = (__bf16*)take((size_t)HD * HD * 2);
  __bf16* WvT = (__bf16*)take((size_t)HD * HD * 2);
  __bf16* WoT = (__bf16*)take((size_t)HD * HD * 2);
  __bf16* W1T = (__bf16*)take((size_t)FD * HD * 2);
  __bf16* W2T = (__bf16*)take((size_t)HD * FD * 2);

  dim3 tb(256);
  dim3 g768(HD / 64, MTOT / 256);
  dim3 g3072(FD / 64, MTOT / 256);
  int vtot = NB * NHEAD * DHEAD * SPAD;

  auto run_block = [&](const float* Wq, const float* bq, const float* Wk, const float* bk,
                       const float* Wv, const float* bv, const float* Wo, const float* bo,
                       const float* W1, const float* b1, const float* W2, const float* b2,
                       const float* l1s, const float* l1b, const float* l2s, const float* l2b,
                       int Sact) {
    dt_wtrans_kernel<<<dim3(HD / 32, HD / 32), tb, 0, stream>>>(Wq, WqT, HD, HD);
    dt_wtrans_kernel<<<dim3(HD / 32, HD / 32), tb, 0, stream>>>(Wk, WkT, HD, HD);
    dt_wtrans_kernel<<<dim3(HD / 32, HD / 32), tb, 0, stream>>>(Wv, WvT, HD, HD);
    dt_wtrans_kernel<<<dim3(HD / 32, HD / 32), tb, 0, stream>>>(Wo, WoT, HD, HD);
    dt_wtrans_kernel<<<dim3(FD / 32, HD / 32), tb, 0, stream>>>(W1, W1T, HD, FD);
    dt_wtrans_kernel<<<dim3(HD / 32, FD / 32), tb, 0, stream>>>(W2, W2T, FD, HD);
    // q,k,v projections
    dt_gemm_kernel<<<g768, tb, 0, stream>>>(hB, WqT, bq, nullptr, nullptr, qB, MTOT, HD, HD, 0);
    dt_gemm_kernel<<<g768, tb, 0, stream>>>(hB, WkT, bk, nullptr, nullptr, kB, MTOT, HD, HD, 0);
    dt_gemm_kernel<<<g768, tb, 0, stream>>>(hB, WvT, bv, nullptr, nullptr, vB, MTOT, HD, HD, 0);
    dt_vtrans_kernel<<<(vtot + 255) / 256, tb, 0, stream>>>(vB, VT);
    // attention (writes back into qB region; safe: each WG reads only what it writes)
    dt_attn_kernel<<<dim3(SPAD / 16, NHEAD, NB), dim3(32), 0, stream>>>(qB, kB, VT, qB, Sact);
    // output proj + residual, then LN1
    dt_gemm_kernel<<<g768, tb, 0, stream>>>(qB, WoT, bo, hF, resF, nullptr, MTOT, HD, HD, 0);
    dt_ln_kernel<<<MTOT, tb, 0, stream>>>(resF, l1s, l1b, hF, hB);
    // MLP: gelu(x@W1+b1)@W2 + b2 + x, then LN2
    dt_gemm_kernel<<<g3072, tb, 0, stream>>>(hB, W1T, b1, nullptr, nullptr, ffB, MTOT, FD, HD, 1);
    dt_gemm_kernel<<<g768, tb, 0, stream>>>(ffB, W2T, b2, hF, resF, nullptr, MTOT, HD, FD, 0);
    dt_ln_kernel<<<MTOT, tb, 0, stream>>>(resF, l2s, l2b, hF, hB);
  };

  // embeddings + ln0
  dt_embed_kernel<<<NB * NT, tb, 0, stream>>>(states, actions, rtg, tsteps, smean, sstd,
                                              Wt, Ws, bs, Wa, ba, Wr, br, hF);
  dt_pad_kernel<<<dim3(SPAD - 3 * NT, NB), tb, 0, stream>>>(hF);
  dt_ln_kernel<<<MTOT, tb, 0, stream>>>(hF, ln0s, ln0b, hF, hB);

  // 6 main blocks (384 valid tokens)
  for (int l = 0; l < 6; l++) {
    run_block(bWq + (size_t)l * HD * HD, bbq + (size_t)l * HD,
              bWk + (size_t)l * HD * HD, bbk + (size_t)l * HD,
              bWv + (size_t)l * HD * HD, bbv + (size_t)l * HD,
              bWo + (size_t)l * HD * HD, bbo + (size_t)l * HD,
              bW1 + (size_t)l * HD * FD, bb1 + (size_t)l * FD,
              bW2 + (size_t)l * FD * HD, bb2 + (size_t)l * HD,
              bl1s + (size_t)l * HD, bl1b + (size_t)l * HD,
              bl2s + (size_t)l * HD, bl2b + (size_t)l * HD, 3 * NT);
  }
  // 2 expert blocks (404 valid tokens: 384 + 20 prompt tokens)
  for (int e = 0; e < 2; e++) {
    dt_tok_kernel<<<dim3(SPAD - 3 * NT, NB), tb, 0, stream>>>(
        eTok + (size_t)e * NTASK * TOKN * HD, task_id, hF, hB);
    run_block(eWq + (size_t)e * HD * HD, ebq + (size_t)e * HD,
              eWk + (size_t)e * HD * HD, ebk + (size_t)e * HD,
              eWv + (size_t)e * HD * HD, ebv + (size_t)e * HD,
              eWo + (size_t)e * HD * HD, ebo + (size_t)e * HD,
              eW1 + (size_t)e * HD * FD, eb1 + (size_t)e * FD,
              eW2 + (size_t)e * FD * HD, eb2 + (size_t)e * HD,
              el1s + (size_t)e * HD, el1b + (size_t)e * HD,
              el2s + (size_t)e * HD, el2b + (size_t)e * HD, 3 * NT + TOKN);
  }
  // heads -> (sp, a, rp) concatenated in d_out
  dt_head_kernel<<<NB * NT, dim3(32), 0, stream>>>(hF, Wpa, bpa, Wps, bps, Wpr, bpr,
                                                   alow, ahigh, (float*)d_out);
}
